// DecomposedRotateNet_18975165514011
// MI455X (gfx1250) — matrix operand order repaired and checked by reading, not compiled, not run on hardware
//
#include <hip/hip_runtime.h>

// ---------------------------------------------------------------------------
// DecomposedRotateNet for MI455X (gfx1250, wave32, WMMA bf16 16x16x32)
//
// B=4096, BITS=64, HID=512.  Dominant cost: [B*64,512]@[512,512] bf16 GEMM.
// All GEMMs run on v_wmma_f32_16x16x32_bf16; LN/softmax/bias epilogues in f32.
// Weights are pre-swizzled once into per-lane-contiguous B-fragment order.
// k_main uses a 32-row M tile per wave so each B fragment feeds 2 wmma.
// ---------------------------------------------------------------------------

#define BATCH 4096
#define BITS  64
#define HID   512
#define LN_EPS 1e-5f

typedef __bf16 bf16_t;
typedef bf16_t v16bf __attribute__((ext_vector_type(16)));
typedef bf16_t v8bf  __attribute__((ext_vector_type(8)));
typedef float  v8f   __attribute__((ext_vector_type(8)));

static __device__ __forceinline__ bf16_t f2bf(float f) {
  unsigned u = __builtin_bit_cast(unsigned, f);
  u += 0x7FFFu + ((u >> 16) & 1u);           // round-to-nearest-even
  unsigned short h = (unsigned short)(u >> 16);
  return __builtin_bit_cast(bf16_t, h);
}

// --- WMMA fragment helpers --------------------------------------------------
// A (16xK bf16, row-major in LDS, leading dim = lda elements):
//   lane<16 : row=lane,   elems 0..7 = K[kc*32+0..7],  8..15 = K[kc*32+16..23]
//   lane>=16: row=lane-16,elems 0..7 = K[kc*32+8..15], 8..15 = K[kc*32+24..31]
static __device__ __forceinline__ v16bf load_a_frag(const bf16_t* lds, int lda, int kc) {
  const int l  = threadIdx.x & 31;
  const int r  = l & 15;
  const int hi = l >> 4;
  const bf16_t* p0 = lds + r * lda + kc * 32 + hi * 8;
  v8bf lo = *(const v8bf*)p0;
  v8bf hb = *(const v8bf*)(p0 + 16);
  return __builtin_shufflevector(lo, hb, 0,1,2,3,4,5,6,7,8,9,10,11,12,13,14,15);
}

// B (32x16 bf16): lane<16 col=lane holds K 0..7,16..23 of the chunk, lane>=16 col=lane-16
// holds K 8..15,24..31.  Stored pre-swizzled so each lane reads 32 contiguous bytes.
static __device__ __forceinline__ v16bf load_b_frag(const bf16_t* F, int KC, int ntile, int kc) {
  const int l = threadIdx.x & 31;
  const bf16_t* p = F + ((size_t)(ntile * KC + kc) * 32 + l) * 16;
  return *(const v16bf*)p;
}

// C/D f32 16x16: lane<16 -> (M=v, N=lane); lane>=16 -> (M=v+8, N=lane-16)

// ---------------------------------------------------------------------------
// Prep: swizzle weight W[n][k] (used as X @ W.T, i.e. B[k][n] = W[n*ld+off+k])
// into fragment order F[((ntile*KC+kc)*32 + lane)*16 + i]
//   with n = ntile*16 + (lane&15), k = kc*32 + (lane>>4)*8 + (i>>3)*16 + (i&7)
// ---------------------------------------------------------------------------
__global__ __launch_bounds__(256)
void k_prep(const float* __restrict__ src, bf16_t* __restrict__ dst,
            int KC, int ld, int col_off, int total) {
  int f = blockIdx.x * 256 + threadIdx.x;
  if (f >= total) return;
  int i   = f & 15;
  int l   = (f >> 4) & 31;
  int blk = f >> 9;
  int kc  = blk % KC;
  int nt  = blk / KC;
  int n   = nt * 16 + (l & 15);
  int k   = kc * 32 + (l >> 4) * 8 + (i >> 3) * 16 + (i & 7);
  dst[f] = f2bf(src[(size_t)n * ld + col_off + k]);
}

// ---------------------------------------------------------------------------
// Shift decoder (fused): X -> L1 -> LN -> ReLU -> L2 -> LN -> ReLU -> L3
//                        -> softmax -> @Wi1[:,64:].T + bi1  => S [B,512] f32
// One wave (32 threads) per block, 16 rows of B per wave.
// ---------------------------------------------------------------------------
__global__ __launch_bounds__(32)
void k_shift(const float* __restrict__ shift_bits,
             const float* __restrict__ b1, const float* __restrict__ g1, const float* __restrict__ be1,
             const float* __restrict__ b2, const float* __restrict__ g2, const float* __restrict__ be2,
             const float* __restrict__ b3,
             const float* __restrict__ bi1,
             const bf16_t* __restrict__ W1F,   // KC=2 , N=512
             const bf16_t* __restrict__ W2F,   // KC=16, N=512
             const bf16_t* __restrict__ W3F,   // KC=16, N=64
             const bf16_t* __restrict__ Wi1sF, // KC=2 , N=512
             float* __restrict__ S) {
  __shared__ float  hbuf[16 * HID];   // 32 KB f32 pre-LN activations
  __shared__ bf16_t abuf[16 * HID];   // 16 KB bf16 A-staging
  __shared__ float  smbuf[16 * BITS]; // 4 KB logits / softmax
  __shared__ float  stat[32];         // mean / rstd per row

  const int tid  = threadIdx.x;
  const int lane = tid & 31;
  const int nl   = lane & 15;
  const int hi   = lane >> 4;
  const int row0 = blockIdx.x * 16;

  // stage X = shift_bits tile as bf16 [16][64]
  for (int i = tid; i < 16 * BITS; i += 32) {
    int r = i >> 6, c = i & 63;
    abuf[i] = f2bf(shift_bits[(size_t)(row0 + r) * BITS + c]);
  }
  __syncthreads();

  // ---- layer 1: [16,64] @ W1F -> hbuf (f32, +b1) ----
  for (int nt = 0; nt < HID / 16; ++nt) {
    v8f c = {};
#pragma unroll
    for (int kc = 0; kc < 2; ++kc) {
      v16bf a = load_a_frag(abuf, BITS, kc);
      v16bf b = load_b_frag(W1F, 2, nt, kc);
      c = __builtin_amdgcn_wmma_f32_16x16x32_bf16(false, a, false, b, (short)0, c, false, false);
    }
    float bias = b1[nt * 16 + nl];
#pragma unroll
    for (int v = 0; v < 8; ++v)
      hbuf[(v + hi * 8) * HID + nt * 16 + nl] = c[v] + bias;
  }
  __syncthreads();

  // ---- LN1 + ReLU -> abuf bf16 ----
  if (lane < 16) {
    float m = 0.f;
    for (int h = 0; h < HID; ++h) m += hbuf[lane * HID + h];
    m *= (1.0f / HID);
    float vv = 0.f;
    for (int h = 0; h < HID; ++h) { float d = hbuf[lane * HID + h] - m; vv += d * d; }
    stat[lane]      = m;
    stat[lane + 16] = __frsqrt_rn(vv * (1.0f / HID) + LN_EPS);
  }
  __syncthreads();
  for (int i = tid; i < 16 * HID; i += 32) {
    int r = i >> 9, h = i & 511;
    float x = (hbuf[i] - stat[r]) * stat[r + 16] * g1[h] + be1[h];
    abuf[i] = f2bf(fmaxf(x, 0.f));
  }
  __syncthreads();

  // ---- layer 2: [16,512] @ W2F -> hbuf (f32, +b2) ----
  for (int nt = 0; nt < HID / 16; ++nt) {
    v8f c = {};
#pragma unroll
    for (int kc = 0; kc < 16; ++kc) {
      v16bf a = load_a_frag(abuf, HID, kc);
      v16bf b = load_b_frag(W2F, 16, nt, kc);
      c = __builtin_amdgcn_wmma_f32_16x16x32_bf16(false, a, false, b, (short)0, c, false, false);
    }
    float bias = b2[nt * 16 + nl];
#pragma unroll
    for (int v = 0; v < 8; ++v)
      hbuf[(v + hi * 8) * HID + nt * 16 + nl] = c[v] + bias;
  }
  __syncthreads();

  // ---- LN2 + ReLU -> abuf bf16 ----
  if (lane < 16) {
    float m = 0.f;
    for (int h = 0; h < HID; ++h) m += hbuf[lane * HID + h];
    m *= (1.0f / HID);
    float vv = 0.f;
    for (int h = 0; h < HID; ++h) { float d = hbuf[lane * HID + h] - m; vv += d * d; }
    stat[lane]      = m;
    stat[lane + 16] = __frsqrt_rn(vv * (1.0f / HID) + LN_EPS);
  }
  __syncthreads();
  for (int i = tid; i < 16 * HID; i += 32) {
    int r = i >> 9, h = i & 511;
    float x = (hbuf[i] - stat[r]) * stat[r + 16] * g2[h] + be2[h];
    abuf[i] = f2bf(fmaxf(x, 0.f));
  }
  __syncthreads();

  // ---- layer 3: [16,512] @ W3F -> smbuf (logits [16,64], +b3) ----
  for (int nt = 0; nt < BITS / 16; ++nt) {
    v8f c = {};
#pragma unroll
    for (int kc = 0; kc < 16; ++kc) {
      v16bf a = load_a_frag(abuf, HID, kc);
      v16bf b = load_b_frag(W3F, 16, nt, kc);
      c = __builtin_amdgcn_wmma_f32_16x16x32_bf16(false, a, false, b, (short)0, c, false, false);
    }
    float bias = b3[nt * 16 + nl];
#pragma unroll
    for (int v = 0; v < 8; ++v)
      smbuf[(v + hi * 8) * BITS + nt * 16 + nl] = c[v] + bias;
  }
  __syncthreads();

  // ---- softmax rows -> abuf bf16 [16][64] ----
  if (lane < 16) {
    float mx = -3.4e38f;
    for (int j = 0; j < BITS; ++j) mx = fmaxf(mx, smbuf[lane * BITS + j]);
    float s = 0.f;
    for (int j = 0; j < BITS; ++j) { float e = __expf(smbuf[lane * BITS + j] - mx); smbuf[lane * BITS + j] = e; s += e; }
    float inv = 1.0f / s;
    for (int j = 0; j < BITS; ++j) smbuf[lane * BITS + j] *= inv;
  }
  __syncthreads();
  for (int i = tid; i < 16 * BITS; i += 32) abuf[i] = f2bf(smbuf[i]);
  __syncthreads();

  // ---- shift_part: [16,64] @ Wi1sF + bi1 -> S (f32, global) ----
  for (int nt = 0; nt < HID / 16; ++nt) {
    v8f c = {};
#pragma unroll
    for (int kc = 0; kc < 2; ++kc) {
      v16bf a = load_a_frag(abuf, BITS, kc);
      v16bf b = load_b_frag(Wi1sF, 2, nt, kc);
      c = __builtin_amdgcn_wmma_f32_16x16x32_bf16(false, a, false, b, (short)0, c, false, false);
    }
    float bias = bi1[nt * 16 + nl];
#pragma unroll
    for (int v = 0; v < 8; ++v)
      S[(size_t)(row0 + v + hi * 8) * HID + nt * 16 + nl] = c[v] + bias;
  }
}

// ---------------------------------------------------------------------------
// Main fused index-net kernel.  One wave per block handles a 32-row tile of
// the flattened M = B*64 dimension (all rows share one b; p = p0..p0+31):
//   h1 = relu(Wi1[h][p] + S[b][h])  (regenerated, never hits HBM)
//   h2 = relu(h1 @ Wi2.T + bi2)     (1024 wmma / wave, 2 per B fragment)
//   logits = h2 @ Wi3.T + bi3 ; softmax ; out = soft @ a_bits[b]
// LDS: h1 32KB + h2 32KB; softmax buffer aliases h1 (dead after GEMM1).
// ---------------------------------------------------------------------------
__global__ __launch_bounds__(32)
void k_main(const float* __restrict__ a_bits,
            const float* __restrict__ Wi1,   // [512,128] row-major (pos part = cols 0..63)
            const float* __restrict__ S,     // [B,512] f32 (shift_part + bi1)
            const float* __restrict__ bi2,
            const float* __restrict__ bi3,
            const bf16_t* __restrict__ Wi2F, // KC=16, N=512
            const bf16_t* __restrict__ Wi3F, // KC=16, N=64
            float* __restrict__ out) {
  __shared__ __align__(16) char smem[65536];
  bf16_t* a1    = (bf16_t*)smem;            // 32 KB: h1 bf16 [32][512]
  bf16_t* a2    = (bf16_t*)(smem + 32768);  // 32 KB: h2 bf16 [32][512]
  float*  smbuf = (float*)smem;             // 8 KB: logits [32][64] (aliases a1)

  const int tid  = threadIdx.x;
  const int lane = tid & 31;
  const int nl   = lane & 15;
  const int hi   = lane >> 4;
  const int m0   = blockIdx.x * 32;
  const int b    = m0 >> 6;       // batch index (32-row tile sits inside one b)
  const int p0   = m0 & 63;       // output-position base (0 or 32)

  // ---- build h1 tile: relu(pos[p] + S[b]) -> bf16 LDS [32][512] ----
  const float* Srow = S + (size_t)b * HID;
  for (int i = tid; i < 32 * HID; i += 32) {
    int r = i >> 9, h = i & 511;
    float x = Wi1[(size_t)h * (2 * BITS) + p0 + r] + Srow[h];
    a1[i] = f2bf(fmaxf(x, 0.f));
  }
  __syncthreads();

  // ---- GEMM 1: h2 = relu(h1 @ Wi2.T + bi2); each B fragment feeds 2 wmma ----
  for (int nt = 0; nt < HID / 16; ++nt) {
    v8f c0 = {}, c1 = {};
#pragma unroll
    for (int kc = 0; kc < 16; ++kc) {
      v16bf bfr = load_b_frag(Wi2F, 16, nt, kc);
      v16bf aa0 = load_a_frag(a1, HID, kc);            // rows 0..15
      v16bf aa1 = load_a_frag(a1 + 16 * HID, HID, kc); // rows 16..31
      c0 = __builtin_amdgcn_wmma_f32_16x16x32_bf16(false, aa0, false, bfr, (short)0, c0, false, false);
      c1 = __builtin_amdgcn_wmma_f32_16x16x32_bf16(false, aa1, false, bfr, (short)0, c1, false, false);
    }
    float bias = bi2[nt * 16 + nl];
#pragma unroll
    for (int v = 0; v < 8; ++v) {
      a2[(v + hi * 8) * HID + nt * 16 + nl]      = f2bf(fmaxf(c0[v] + bias, 0.f));
      a2[(16 + v + hi * 8) * HID + nt * 16 + nl] = f2bf(fmaxf(c1[v] + bias, 0.f));
    }
  }
  __syncthreads();   // a2 ready; a1 region now dead -> reused as smbuf

  // ---- GEMM 2: logits = h2 @ Wi3.T + bi3 -> smbuf [32][64] ----
  for (int nt = 0; nt < BITS / 16; ++nt) {
    v8f c0 = {}, c1 = {};
#pragma unroll
    for (int kc = 0; kc < 16; ++kc) {
      v16bf bfr = load_b_frag(Wi3F, 16, nt, kc);
      v16bf aa0 = load_a_frag(a2, HID, kc);
      v16bf aa1 = load_a_frag(a2 + 16 * HID, HID, kc);
      c0 = __builtin_amdgcn_wmma_f32_16x16x32_bf16(false, aa0, false, bfr, (short)0, c0, false, false);
      c1 = __builtin_amdgcn_wmma_f32_16x16x32_bf16(false, aa1, false, bfr, (short)0, c1, false, false);
    }
    float bias = bi3[nt * 16 + nl];
#pragma unroll
    for (int v = 0; v < 8; ++v) {
      smbuf[(v + hi * 8) * BITS + nt * 16 + nl]      = c0[v] + bias;
      smbuf[(16 + v + hi * 8) * BITS + nt * 16 + nl] = c1[v] + bias;
    }
  }
  __syncthreads();

  // ---- softmax + weighted gather against a_bits[b]; lane r handles row r ----
  {
    const int r = lane;  // 32 rows, 32 lanes
    float mx = -3.4e38f;
    for (int j = 0; j < BITS; ++j) mx = fmaxf(mx, smbuf[r * BITS + j]);
    float s = 0.f;
    for (int j = 0; j < BITS; ++j) { float e = __expf(smbuf[r * BITS + j] - mx); smbuf[r * BITS + j] = e; s += e; }
    float inv = 1.0f / s;
    const float* arow = a_bits + (size_t)b * BITS;
    float acc = 0.f;
    for (int j = 0; j < BITS; ++j) acc += smbuf[r * BITS + j] * arow[j];
    out[(size_t)m0 + r] = acc * inv;
  }
}

// ---------------------------------------------------------------------------
// Launch
// ---------------------------------------------------------------------------
extern "C" void kernel_launch(void* const* d_in, const int* in_sizes, int n_in,
                              void* d_out, int out_size, void* d_ws, size_t ws_size,
                              hipStream_t stream) {
  const float* a_bits     = (const float*)d_in[0];
  const float* shift_bits = (const float*)d_in[1];
  const float* W1  = (const float*)d_in[2];
  const float* b1  = (const float*)d_in[3];
  const float* g1  = (const float*)d_in[4];
  const float* be1 = (const float*)d_in[5];
  const float* W2  = (const float*)d_in[6];
  const float* b2  = (const float*)d_in[7];
  const float* g2  = (const float*)d_in[8];
  const float* be2 = (const float*)d_in[9];
  const float* W3  = (const float*)d_in[10];
  const float* b3  = (const float*)d_in[11];
  const float* Wi1 = (const float*)d_in[12];
  const float* bi1 = (const float*)d_in[13];
  const float* Wi2 = (const float*)d_in[14];
  const float* bi2 = (const float*)d_in[15];
  const float* Wi3 = (const float*)d_in[16];
  const float* bi3 = (const float*)d_in[17];
  float* out = (float*)d_out;

  // workspace layout (bytes)
  char* ws = (char*)d_ws;
  float*  S     = (float*)ws;                                  // 4096*512*4 = 8 MiB
  size_t  off   = (size_t)BATCH * HID * sizeof(float);
  bf16_t* W1F   = (bf16_t*)(ws + off); off += (size_t)32768  * 2;  // N=512,K=64
  bf16_t* W2F   = (bf16_t*)(ws + off); off += (size_t)262144 * 2;  // N=512,K=512
  bf16_t* W3F   = (bf16_t*)(ws + off); off += (size_t)32768  * 2;  // N=64 ,K=512
  bf16_t* Wi1sF = (bf16_t*)(ws + off); off += (size_t)32768  * 2;  // N=512,K=64
  bf16_t* Wi2F  = (bf16_t*)(ws + off); off += (size_t)262144 * 2;  // N=512,K=512
  bf16_t* Wi3F  = (bf16_t*)(ws + off); off += (size_t)32768  * 2;  // N=64 ,K=512

  // weight swizzle+convert (once per call; cheap, weights stay in L2)
  k_prep<<<32768  / 256, 256, 0, stream>>>(W1,  W1F,   2,  BITS,     0, 32768);
  k_prep<<<262144 / 256, 256, 0, stream>>>(W2,  W2F,   16, HID,      0, 262144);
  k_prep<<<32768  / 256, 256, 0, stream>>>(W3,  W3F,   16, HID,      0, 32768);
  k_prep<<<32768  / 256, 256, 0, stream>>>(Wi1, Wi1sF, 2,  2 * BITS, BITS, 32768);
  k_prep<<<262144 / 256, 256, 0, stream>>>(Wi2, Wi2F,  16, HID,      0, 262144);
  k_prep<<<32768  / 256, 256, 0, stream>>>(Wi3, Wi3F,  16, HID,      0, 32768);

  // fused shift decoder + shift_part  -> S
  k_shift<<<BATCH / 16, 32, 0, stream>>>(shift_bits, b1, g1, be1, b2, g2, be2, b3,
                                         bi1, W1F, W2F, W3F, Wi1sF, S);

  // fused index net + softmax-gather -> out (32-row tiles)
  k_main<<<(BATCH * BITS) / 32, 32, 0, stream>>>(a_bits, Wi1, S, bi2, bi3,
                                                 Wi2F, Wi3F, out);
}
